// GCN_raw_att_46729244181071
// MI455X (gfx1250) — matrix operand order repaired and checked by reading, not compiled, not run on hardware
//
#include <hip/hip_runtime.h>
#include <hip/hip_bf16.h>

typedef __attribute__((ext_vector_type(16))) __bf16   v16bf;
typedef __attribute__((ext_vector_type(8)))  float    v8f;
typedef __attribute__((ext_vector_type(4)))  unsigned uint4v;
typedef __attribute__((ext_vector_type(8)))  int      int8v;
typedef __attribute__((ext_vector_type(4)))  int      int4v;

#define NNODES 100000
#define NGRAPHS 1024
#define BN_EPS 1e-5f

// ---------------------------------------------------------------- utilities
__global__ void k_zero(float* __restrict__ p, int n) {
    int i = blockIdx.x * blockDim.x + threadIdx.x;
    int stride = gridDim.x * blockDim.x;
    for (; i < n; i += stride) p[i] = 0.0f;
}

__global__ void k_deg(const int* __restrict__ dst, float* __restrict__ deg, int nE) {
    int e = blockIdx.x * blockDim.x + threadIdx.x;
    if (e < nE) atomicAdd(&deg[dst[e]], 1.0f);
}

// message scatter: s[dst] += ew * x[src]   (segment_sum of weighted messages)
template<int F>
__global__ void k_scatter(const float* __restrict__ x, const int* __restrict__ src,
                          const int* __restrict__ dst, const float* __restrict__ ew,
                          float* __restrict__ s, int nE) {
    int e = blockIdx.x * blockDim.x + threadIdx.x;
    if (e >= nE) return;
    int sn = src[e], dn = dst[e];
    float w = ew[e];
    if (e + 256 < nE) {
        // stream-ahead prefetch of the next gather row -> global_prefetch_b8
        __builtin_prefetch(&x[src[e + 256] * F], 0, 1);
    }
    const float* xs = &x[(size_t)sn * F];
    float*       sd = &s[(size_t)dn * F];
#pragma unroll
    for (int f = 0; f < F; ++f) atomicAdd(&sd[f], xs[f] * w);
}

// ------------------------------------------------- TDM weight staging helper
// Build a 2-D Tensor DMA Descriptor (D#) per CDNA5 ISA
// and issue TENSOR_LOAD_TO_LDS: rows x cols fp32 tile, row stride = cols.
__device__ __forceinline__ unsigned lds_off_of(const void* p) {
    // LDS aperture: flat addr low dword is the LDS byte offset (ISA 10.2)
    return (unsigned)(uintptr_t)p;
}

__device__ __forceinline__ void tdm_load_2d(unsigned lds_off, const float* gsrc,
                                            int rows, int cols) {
#if __has_builtin(__builtin_amdgcn_tensor_load_to_lds)
    unsigned long long ga = (unsigned long long)(uintptr_t)gsrc;
    uint4v g0 = {};
    g0[0] = 1u;                                            // count=1, user mode
    g0[1] = lds_off;                                       // lds_addr (bytes)
    g0[2] = (unsigned)(ga & 0xFFFFFFFFu);                  // global_addr[31:0]
    g0[3] = (unsigned)((ga >> 32) & 0x1FFFFFFu)            // global_addr[56:32]
          | (2u << 30);                                    // type = 2 (image)
    int8v g1 = {};
    unsigned td0 = (unsigned)cols;                         // dim0 = row length
    unsigned td1 = (unsigned)rows;
    g1[0] = 0x20000;                                       // data_size=2 (4B), wg_mask=0
    g1[1] = (int)((td0 & 0xFFFFu) << 16);                  // tensor_dim0[15:0] @ bit48
    g1[2] = (int)(((td0 >> 16) & 0xFFFFu) |                // tensor_dim0[31:16]
                  ((td1 & 0xFFFFu) << 16));                // tensor_dim1[15:0]
    g1[3] = (int)(((td1 >> 16) & 0xFFFFu) |                // tensor_dim1[31:16]
                  ((td0 & 0xFFFFu) << 16));                // tile_dim0 @ [127:112]
    g1[4] = (int)(td1 & 0xFFFFu);                          // tile_dim1; tile_dim2=0
    g1[5] = (int)td0;                                      // tensor_dim0_stride[31:0]
    g1[6] = 0;                                             // stride hi, dim1_stride lo
    g1[7] = 0;
    int4v g2 = {}, g3 = {};
#if defined(__clang_major__) && __clang_major__ >= 23
    int8v g4 = {};
    __builtin_amdgcn_tensor_load_to_lds(g0, g1, g2, g3, g4, 0);
#else
    __builtin_amdgcn_tensor_load_to_lds(g0, g1, g2, g3, 0);
#endif
#endif
}

// ---------------------------------------------------------------- WMMA linear
// out[n,FOUT] = (MEAN ? in/deg : in) @ w (+ bias) (+ root @ wroot) (ReLU opt.)
// One wave per 16-row tile; K padded/looped to multiples of 32 (bf16 WMMA).
// Weights staged into LDS once per block via the Tensor Data Mover.
template<int FIN, int FOUT, bool MEAN, bool ROOT, bool RELU>
__global__ void k_linear_wmma(const float* __restrict__ in,
                              const float* __restrict__ deg,
                              const float* __restrict__ root,
                              const float* __restrict__ w,
                              const float* __restrict__ bias,
                              const float* __restrict__ wroot,
                              float* __restrict__ out, int nrows) {
    __shared__ float lw[FIN * FOUT];
    __shared__ float lwr[ROOT ? FIN * FOUT : 16];

#if __has_builtin(__builtin_amdgcn_tensor_load_to_lds)
    if (threadIdx.x == 0) {                     // TDM ignores EXEC; one issue/block
        tdm_load_2d(lds_off_of(&lw[0]), w, FIN, FOUT);
        if (ROOT) tdm_load_2d(lds_off_of(&lwr[0]), wroot, FIN, FOUT);
        __builtin_amdgcn_s_wait_tensorcnt(0);
    }
#else
    for (int i = threadIdx.x; i < FIN * FOUT; i += blockDim.x) {
        lw[i] = w[i];
        if (ROOT) lwr[i] = wroot[i];
    }
#endif
    __syncthreads();

    const int lane = threadIdx.x & 31;
    const int wid  = threadIdx.x >> 5;
    const int tile = blockIdx.x * 8 + wid;          // 8 waves / 256-thread block
    const int ntiles = nrows >> 4;
    if (tile >= ntiles) return;                     // wave-uniform: EXEC stays full
    const int mrow = lane & 15;                     // A-row within tile
    const int col  = lane & 15;                     // B/D column within tile
    const int hi   = lane >> 4;                     // lane-half selects K range
    const int row  = tile * 16 + mrow;
    float inv = 1.0f;
    if constexpr (MEAN) inv = 1.0f / fmaxf(deg[row], 1.0f);
    constexpr int KT = (FIN + 31) / 32;

#pragma unroll
    for (int ct = 0; ct < FOUT / 16; ++ct) {
        v8f acc;
        float bv = bias ? bias[ct * 16 + col] : 0.0f;
#pragma unroll
        for (int r = 0; r < 8; ++r) acc[r] = bv;

#pragma unroll
        for (int kt = 0; kt < KT; ++kt) {
            v16bf a, b;
#pragma unroll
            for (int e2 = 0; e2 < 16; ++e2) {       // A frag: 16x32 bf16 layout
                int k = kt * 32 + 16 * hi + e2;
                float av = (k < FIN) ? in[(size_t)row * FIN + k] * inv : 0.0f;
                a[e2] = (__bf16)av;
            }
#pragma unroll
            for (int e2 = 0; e2 < 16; ++e2) {       // B frag: 32x16 bf16 (from LDS)
                int k = kt * 32 + 16 * hi + e2;
                float bvv = (k < FIN) ? lw[k * FOUT + ct * 16 + col] : 0.0f;
                b[e2] = (__bf16)bvv;
            }
            acc = __builtin_amdgcn_wmma_f32_16x16x32_bf16(
                false, a, false, b, (short)0, acc, false, false);

            if constexpr (ROOT) {
                v16bf ar, br;
#pragma unroll
                for (int e2 = 0; e2 < 16; ++e2) {
                    int k = kt * 32 + 16 * hi + e2;
                    ar[e2] = (__bf16)((k < FIN) ? root[(size_t)row * FIN + k] : 0.0f);
                }
#pragma unroll
                for (int e2 = 0; e2 < 16; ++e2) {
                    int k = kt * 32 + 16 * hi + e2;
                    br[e2] = (__bf16)((k < FIN) ? lwr[k * FOUT + ct * 16 + col] : 0.0f);
                }
                acc = __builtin_amdgcn_wmma_f32_16x16x32_bf16(
                    false, ar, false, br, (short)0, acc, false, false);
            }
        }
#pragma unroll
        for (int r = 0; r < 8; ++r) {               // D frag: M = r + 8*hi
            float v = acc[r];
            if (RELU) v = fmaxf(v, 0.0f);
            out[(size_t)(tile * 16 + r + 8 * hi) * FOUT + ct * 16 + col] = v;
        }
    }
}

// ---------------------------------------------------------------- batch norm
// stats layout: [0,64) sum  [64,128) sumsq  [128,192) scale  [192,256) shift
template<int F>
__global__ void k_bn_stats(const float* __restrict__ h, float* __restrict__ stats, int n) {
    __shared__ float ls[F], lq[F];
    for (int j = threadIdx.x; j < F; j += blockDim.x) { ls[j] = 0.f; lq[j] = 0.f; }
    __syncthreads();
    int total = n * F;
    for (int i = blockIdx.x * blockDim.x + threadIdx.x; i < total;
         i += gridDim.x * blockDim.x) {
        float v = h[i];
        int f = i & (F - 1);
        atomicAdd(&ls[f], v);
        atomicAdd(&lq[f], v * v);
    }
    __syncthreads();
    for (int j = threadIdx.x; j < F; j += blockDim.x) {
        atomicAdd(&stats[j], ls[j]);
        atomicAdd(&stats[64 + j], lq[j]);
    }
}

template<int F>
__global__ void k_bn_finalize(const float* __restrict__ g, const float* __restrict__ b,
                              float* __restrict__ stats, int n) {
    int f = threadIdx.x;
    if (f < F) {
        float mu  = stats[f] / (float)n;
        float var = stats[64 + f] / (float)n - mu * mu;   // biased, matches jnp.var
        float sc  = g[f] * rsqrtf(var + BN_EPS);
        stats[128 + f] = sc;
        stats[192 + f] = b[f] - mu * sc;
    }
}

template<int F>
__global__ void k_bn_apply_relu(float* __restrict__ h, const float* __restrict__ stats, int n) {
    int total = n * F;
    for (int i = blockIdx.x * blockDim.x + threadIdx.x; i < total;
         i += gridDim.x * blockDim.x) {
        int f = i & (F - 1);
        h[i] = fmaxf(h[i] * stats[128 + f] + stats[192 + f], 0.0f);
    }
}

// ---------------------------------------------------------------- attention pool
__device__ __forceinline__ unsigned f2ord(float f) {
    unsigned u = __float_as_uint(f);
    return (u & 0x80000000u) ? ~u : (u | 0x80000000u);
}
__device__ __forceinline__ float ord2f(unsigned u) {
    return __uint_as_float((u & 0x80000000u) ? (u & 0x7fffffffu) : ~u);
}

// gate score (32->1) + segment max (order-preserving uint atomicMax)
__global__ void k_gate2(const float* __restrict__ hg, const float* __restrict__ w2,
                        const float* __restrict__ b2, const int* __restrict__ batch,
                        float* __restrict__ gate, unsigned* __restrict__ gmax, int n) {
    int i = blockIdx.x * blockDim.x + threadIdx.x;
    if (i >= n) return;
    float acc = b2[0];
    const float* r = &hg[(size_t)i * 32];
#pragma unroll
    for (int j = 0; j < 32; ++j) acc += r[j] * w2[j];
    gate[i] = acc;
    atomicMax(&gmax[batch[i]], f2ord(acc));
}

__global__ void k_expsum(float* __restrict__ gate, const unsigned* __restrict__ gmax,
                         const int* __restrict__ batch, float* __restrict__ denom, int n) {
    int i = blockIdx.x * blockDim.x + threadIdx.x;
    if (i >= n) return;
    int b = batch[i];
    float e = expf(gate[i] - ord2f(gmax[b]));
    gate[i] = e;
    atomicAdd(&denom[b], e);
}

__global__ void k_pool(const float* __restrict__ h, const float* __restrict__ gate,
                       const float* __restrict__ denom, const int* __restrict__ batch,
                       float* __restrict__ pooled, int n) {
    int idx = blockIdx.x * blockDim.x + threadIdx.x;
    if (idx >= n * 64) return;
    int node = idx >> 6, f = idx & 63;
    int b = batch[node];
    float alpha = gate[node] / denom[b];
    atomicAdd(&pooled[(size_t)b * 64 + f], alpha * h[idx]);
}

// fc4 (32->2) + log_softmax
__global__ void k_head(const float* __restrict__ f1, const float* __restrict__ w4,
                       const float* __restrict__ b4, float* __restrict__ out, int ngr) {
    int g = blockIdx.x * blockDim.x + threadIdx.x;
    if (g >= ngr) return;
    float o0 = b4[0], o1 = b4[1];
    const float* r = &f1[(size_t)g * 32];
#pragma unroll
    for (int j = 0; j < 32; ++j) { o0 += r[j] * w4[j * 2]; o1 += r[j] * w4[j * 2 + 1]; }
    float m = fmaxf(o0, o1);
    float l = logf(expf(o0 - m) + expf(o1 - m));
    out[g * 2]     = o0 - m - l;
    out[g * 2 + 1] = o1 - m - l;
}

// ---------------------------------------------------------------- launch
extern "C" void kernel_launch(void* const* d_in, const int* in_sizes, int n_in,
                              void* d_out, int out_size, void* d_ws, size_t ws_size,
                              hipStream_t stream) {
    const float* x    = (const float*)d_in[0];
    const float* ew   = (const float*)d_in[1];
    const int*   ei   = (const int*)  d_in[2];
    const int*   bat  = (const int*)  d_in[3];
    const float* w1r  = (const float*)d_in[4];
    const float* b1r  = (const float*)d_in[5];
    const float* w1o  = (const float*)d_in[6];
    const float* bn1g = (const float*)d_in[7];
    const float* bn1b = (const float*)d_in[8];
    const float* w2r  = (const float*)d_in[9];
    const float* b2r  = (const float*)d_in[10];
    const float* w2o  = (const float*)d_in[11];
    const float* bn2g = (const float*)d_in[12];
    const float* bn2b = (const float*)d_in[13];
    const float* w3r  = (const float*)d_in[14];
    const float* b3r  = (const float*)d_in[15];
    const float* w3o  = (const float*)d_in[16];
    const float* gw1  = (const float*)d_in[17];
    const float* gb1  = (const float*)d_in[18];
    const float* gw2  = (const float*)d_in[19];
    const float* gb2  = (const float*)d_in[20];
    const float* f1w  = (const float*)d_in[21];
    const float* f1b  = (const float*)d_in[22];
    const float* f4w  = (const float*)d_in[23];
    const float* f4b  = (const float*)d_in[24];

    const int nE = in_sizes[1];
    const int N  = NNODES;
    const int G  = NGRAPHS;
    const int* src = ei;
    const int* dst = ei + nE;

    float* ws     = (float*)d_ws;
    float* S      = ws;                        // N*64 scatter accumulator
    float* A      = S + (size_t)N * 64;        // N*64 ping buffer
    float* B      = A + (size_t)N * 64;        // N*64 pong buffer
    float* deg    = B + (size_t)N * 64;        // N
    float* gate   = deg + N;                   // N
    float* stats  = gate + N;                  // 256
    unsigned* gmax = (unsigned*)(stats + 256); // G
    float* denom  = (float*)(gmax + G);        // G
    float* pooled = denom + G;                 // G*64
    float* fc1o   = pooled + (size_t)G * 64;   // G*32

    const int TB = 256;
    auto nb = [](long n, int tb) { return (int)((n + tb - 1) / tb); };
    const int gdimN = (N / 16 + 7) / 8;        // 6250 tiles, 8 waves/block
    const int gdimG = (G / 16 + 7) / 8;        // 64 tiles

    // degrees (shared by all three convs)
    k_zero<<<nb(N, TB), TB, 0, stream>>>(deg, N);
    k_deg<<<nb(nE, TB), TB, 0, stream>>>(dst, deg, nE);

    // ---- GraphConv 1 (4 -> 16) + BN + ReLU
    k_zero<<<nb((long)N * 4, TB), TB, 0, stream>>>(S, N * 4);
    k_scatter<4><<<nb(nE, TB), TB, 0, stream>>>(x, src, dst, ew, S, nE);
    k_linear_wmma<4, 16, true, true, false><<<gdimN, TB, 0, stream>>>(
        S, deg, x, w1r, b1r, w1o, A, N);
    k_zero<<<1, 128, 0, stream>>>(stats, 128);
    k_bn_stats<16><<<512, TB, 0, stream>>>(A, stats, N);
    k_bn_finalize<16><<<1, 64, 0, stream>>>(bn1g, bn1b, stats, N);
    k_bn_apply_relu<16><<<nb((long)N * 16, TB), TB, 0, stream>>>(A, stats, N);

    // ---- GraphConv 2 (16 -> 32) + BN + ReLU
    k_zero<<<nb((long)N * 16, TB), TB, 0, stream>>>(S, N * 16);
    k_scatter<16><<<nb(nE, TB), TB, 0, stream>>>(A, src, dst, ew, S, nE);
    k_linear_wmma<16, 32, true, true, false><<<gdimN, TB, 0, stream>>>(
        S, deg, A, w2r, b2r, w2o, B, N);
    k_zero<<<1, 128, 0, stream>>>(stats, 128);
    k_bn_stats<32><<<512, TB, 0, stream>>>(B, stats, N);
    k_bn_finalize<32><<<1, 64, 0, stream>>>(bn2g, bn2b, stats, N);
    k_bn_apply_relu<32><<<nb((long)N * 32, TB), TB, 0, stream>>>(B, stats, N);

    // ---- GraphConv 3 (32 -> 64) + ReLU (fused)
    k_zero<<<nb((long)N * 32, TB), TB, 0, stream>>>(S, N * 32);
    k_scatter<32><<<nb(nE, TB), TB, 0, stream>>>(B, src, dst, ew, S, nE);
    k_linear_wmma<32, 64, true, true, true><<<gdimN, TB, 0, stream>>>(
        S, deg, B, w3r, b3r, w3o, A, N);           // A = h3 [N,64]

    // ---- gate hidden (64 -> 32) + ReLU (fused), no mean, no root
    k_linear_wmma<64, 32, false, false, true><<<gdimN, TB, 0, stream>>>(
        A, nullptr, nullptr, gw1, gb1, nullptr, B, N);

    // ---- attention pooling
    k_zero<<<nb((long)(G + G + G * 64), TB), TB, 0, stream>>>((float*)gmax, G + G + G * 64);
    k_gate2<<<nb(N, TB), TB, 0, stream>>>(B, gw2, gb2, bat, gate, gmax, N);
    k_expsum<<<nb(N, TB), TB, 0, stream>>>(gate, gmax, bat, denom, N);
    k_pool<<<nb((long)N * 64, TB), TB, 0, stream>>>(A, gate, denom, bat, pooled, N);

    // ---- fc1 (64 -> 32) + ReLU (WMMA, 64 tiles of 16 graphs)
    k_linear_wmma<64, 32, false, false, true><<<gdimG, TB, 0, stream>>>(
        pooled, nullptr, nullptr, f1w, f1b, nullptr, fc1o, G);

    // ---- fc4 + log_softmax -> d_out [G,2]
    k_head<<<nb(G, TB), TB, 0, stream>>>(fc1o, f4w, f4b, (float*)d_out, G);
}
